// SelfAttentionV2_48627619725379
// MI455X (gfx1250) — matrix operand order repaired and checked by reading, not compiled, not run on hardware
//
#include <hip/hip_runtime.h>

// MI455X / gfx1250 self-attention: bf16 WMMA + async global->LDS staging.
// Pipeline: (1) QKV = X*W + b  -> bf16 workspace (48MB, L2-resident)
//           per batch: (2) S = Q*K^T * 1/32 (f32, 16MB), (3) row softmax ->
//           bf16 P written in place over S, (4) out = P*V (f32 -> d_out).

typedef __bf16 bf16_t;
typedef bf16_t v16bf __attribute__((ext_vector_type(16)));
typedef float  v8f   __attribute__((ext_vector_type(8)));
typedef int    v4i   __attribute__((ext_vector_type(4)));

#define HID  1024
#define NQKV 3072
#define SEQ  2048

static __device__ __forceinline__ v8f wmma_bf16(v16bf a, v16bf b, v8f c) {
  // (neg_a, A, neg_b, B, c_mod, C, reuse_a, reuse_b)
  return __builtin_amdgcn_wmma_f32_16x16x32_bf16(false, a, false, b, (short)0, c,
                                                 false, false);
}

// ---- gfx1250 async global->LDS copy (16B per call), ASYNCcnt-tracked --------
#if __has_builtin(__builtin_amdgcn_global_load_async_to_lds_b128)
typedef __attribute__((address_space(1))) v4i as1_v4i;  // global
typedef __attribute__((address_space(3))) v4i as3_v4i;  // LDS
#define HAVE_ASYNC_LDS 1
#endif

static __device__ __forceinline__ void async_copy16(const void* g, void* l) {
#ifdef HAVE_ASYNC_LDS
  __builtin_amdgcn_global_load_async_to_lds_b128((as1_v4i*)g, (as3_v4i*)l, 0, 0);
#else
  *(uint4*)l = *(const uint4*)g;  // fallback: synchronous 16B copy
#endif
}

static __device__ __forceinline__ void wait_async0() {
#if __has_builtin(__builtin_amdgcn_s_wait_asynccnt)
  __builtin_amdgcn_s_wait_asynccnt(0);
#else
  asm volatile("s_wait_asynccnt 0x0" ::: "memory");
#endif
}

// ---- Direct-from-memory fragment loaders (contiguous k = 16g+i, wave32) ---
// A 16x32 (MxK): lane L -> row m = L&15, group g = L>>4.
//   halves 0..15 -> k = g*16+i
static __device__ __forceinline__ v16bf load_a_bf16(const bf16_t* __restrict__ A,
                                                    int pitch, int row0, int k0,
                                                    int lane) {
  const int m = row0 + (lane & 15);
  const int g = lane >> 4;
  const bf16_t* p = A + (size_t)m * pitch + k0 + g * 16;
  v16bf a;
#pragma unroll
  for (int i = 0; i < 8; ++i) a[i] = p[i];
#pragma unroll
  for (int i = 0; i < 8; ++i) a[i + 8] = p[8 + i];
  return a;
}

// B[k,n] = M[n,k] (M row-major): per-lane reads are 32B-contiguous.
static __device__ __forceinline__ v16bf load_bT_bf16(const bf16_t* __restrict__ M,
                                                     int pitch, int k0, int col0,
                                                     int lane) {
  const int n = col0 + (lane & 15);
  const int g = lane >> 4;
  const bf16_t* p = M + (size_t)n * pitch + k0 + g * 16;
  v16bf b;
#pragma unroll
  for (int i = 0; i < 16; ++i) b[i] = p[i];
  return b;
}

// ---- LDS fragment loaders ---------------------------------------------------
static __device__ __forceinline__ v16bf lds_a_f32(const float* Ab, int row, int g) {
  const float* p = Ab + row * 32 + g * 16;
  v16bf a;
#pragma unroll
  for (int i = 0; i < 8; ++i) a[i] = (bf16_t)p[i];
#pragma unroll
  for (int i = 0; i < 8; ++i) a[i + 8] = (bf16_t)p[8 + i];
  return a;
}
static __device__ __forceinline__ v16bf lds_b_f32(const float* Bb, int n, int g) {
  const float* p = Bb + g * 16 * 128 + n;
  v16bf b;
#pragma unroll
  for (int i = 0; i < 16; ++i) b[i] = (bf16_t)p[i * 128];
  return b;
}
static __device__ __forceinline__ v16bf lds_a_bf16(const bf16_t* Ab, int row, int g) {
  const bf16_t* p = Ab + row * 32 + g * 16;
  v16bf a;
#pragma unroll
  for (int i = 0; i < 8; ++i) a[i] = p[i];
#pragma unroll
  for (int i = 0; i < 8; ++i) a[i + 8] = p[8 + i];
  return a;
}
static __device__ __forceinline__ v16bf lds_b_bf16(const bf16_t* Bb, int n, int g) {
  const bf16_t* p = Bb + g * 16 * 128 + n;
  v16bf b;
#pragma unroll
  for (int i = 0; i < 16; ++i) b[i] = p[i * 128];
  return b;
}

// ---- Kernel 1: QKV = X*W + bias (f32 in, bf16 out) --------------------------
// Block tile 128x128, 8 waves (4x2), wave tile 32x64, BK=32.
// Double-buffered LDS staged with async global->LDS copies.
__global__ void __launch_bounds__(256)
qkv_gemm_kernel(const float* __restrict__ X, const float* __restrict__ W,
                const float* __restrict__ bias, bf16_t* __restrict__ qkv) {
  __shared__ float As[2][128 * 32];  // [row][k], 16KB each
  __shared__ float Bs[2][32 * 128];  // [k][n],  16KB each

  const int tid   = threadIdx.x;
  const int lane  = tid & 31;
  const int wave  = tid >> 5;
  const int waveM = wave >> 1;       // 0..3
  const int waveN = wave & 1;        // 0..1
  const int g     = lane >> 4;
  const int nl    = lane & 15;

  const int nBlkN = NQKV / 128;      // 24
  const int m0 = (blockIdx.x / nBlkN) * 128;
  const int n0 = (blockIdx.x % nBlkN) * 128;

  auto stage = [&](int buf, int k0) {
#pragma unroll
    for (int r = 0; r < 4; ++r) {    // A: 1024 16B chunks (4/thread)
      const int c = tid + r * 256;
      const int row = c >> 3, koff = (c & 7) * 4;
      async_copy16(X + (size_t)(m0 + row) * HID + k0 + koff,
                   &As[buf][row * 32 + koff]);
    }
#pragma unroll
    for (int r = 0; r < 4; ++r) {    // B: 1024 16B chunks (4/thread)
      const int c = tid + r * 256;
      const int kr = c >> 5, noff = (c & 31) * 4;
      async_copy16(W + (size_t)(k0 + kr) * NQKV + n0 + noff,
                   &Bs[buf][kr * 128 + noff]);
    }
  };

  v8f acc[2][4];
#pragma unroll
  for (int s = 0; s < 2; ++s)
#pragma unroll
    for (int t = 0; t < 4; ++t) acc[s][t] = v8f{};

  stage(0, 0);
  int cur = 0;
  for (int kt = 0; kt < HID / 32; ++kt) {
    wait_async0();
    __syncthreads();                 // tile[cur] ready; prev compute finished
    if (kt + 1 < HID / 32) stage(cur ^ 1, (kt + 1) * 32);

    const float* Ab = &As[cur][0];
    const float* Bb = &Bs[cur][0];
    v16bf A0 = lds_a_f32(Ab, waveM * 32 + nl, g);
    v16bf A1 = lds_a_f32(Ab, waveM * 32 + 16 + nl, g);
    v16bf Bt[4];
#pragma unroll
    for (int t = 0; t < 4; ++t) Bt[t] = lds_b_f32(Bb, waveN * 64 + t * 16 + nl, g);
#pragma unroll
    for (int t = 0; t < 4; ++t) {
      acc[0][t] = wmma_bf16(A0, Bt[t], acc[0][t]);
      acc[1][t] = wmma_bf16(A1, Bt[t], acc[1][t]);
    }
    cur ^= 1;
  }

  const int mw = m0 + waveM * 32;
  const int nw = n0 + waveN * 64;
#pragma unroll
  for (int s = 0; s < 2; ++s)
#pragma unroll
    for (int t = 0; t < 4; ++t) {
      const int col = nw + t * 16 + nl;
      const float bv = bias[col];
#pragma unroll
      for (int j = 0; j < 8; ++j) {
        const int row = mw + s * 16 + g * 8 + j;
        qkv[(size_t)row * NQKV + col] = (bf16_t)(acc[s][t][j] + bv);
      }
    }
}

// ---- Kernel 2: S = (Q*K^T) / sqrt(D) for one batch --------------------------
// Per-lane loads are contiguous (Q and K rows 32B); add K-tile prefetch.
__global__ void __launch_bounds__(256)
scores_kernel(const bf16_t* __restrict__ qkv_b, float* __restrict__ S) {
  const int lane = threadIdx.x & 31;
  const int wid  = blockIdx.x * 8 + (threadIdx.x >> 5);
  const int nMacroN = SEQ / 64;                // 32
  const int q0 = (wid / nMacroN) * 32;
  const int n0 = (wid % nMacroN) * 64;

  const bf16_t* Q = qkv_b;                     // cols [0,1024)
  const bf16_t* K = qkv_b + HID;               // cols [1024,2048)

  v8f acc[2][4];
#pragma unroll
  for (int s = 0; s < 2; ++s)
#pragma unroll
    for (int t = 0; t < 4; ++t) acc[s][t] = v8f{};

  for (int k0 = 0; k0 < HID; k0 += 32) {
    if (k0 + 32 < HID) {                       // global_prefetch_b8 next tiles
      __builtin_prefetch(Q + (size_t)(q0 + (lane & 15)) * NQKV + k0 + 32, 0, 0);
      __builtin_prefetch(K + (size_t)(n0 + lane) * NQKV + k0 + 32, 0, 0);
    }
    v16bf A0 = load_a_bf16(Q, NQKV, q0, k0, lane);
    v16bf A1 = load_a_bf16(Q, NQKV, q0 + 16, k0, lane);
    v16bf Bt[4];
#pragma unroll
    for (int t = 0; t < 4; ++t) Bt[t] = load_bT_bf16(K, NQKV, k0, n0 + t * 16, lane);
#pragma unroll
    for (int t = 0; t < 4; ++t) {
      acc[0][t] = wmma_bf16(A0, Bt[t], acc[0][t]);
      acc[1][t] = wmma_bf16(A1, Bt[t], acc[1][t]);
    }
  }

  const float scale = 0.03125f;                // 1/sqrt(1024)
  const int g  = lane >> 4;
  const int nl = lane & 15;
#pragma unroll
  for (int s = 0; s < 2; ++s)
#pragma unroll
    for (int t = 0; t < 4; ++t)
#pragma unroll
      for (int j = 0; j < 8; ++j) {
        const int row = q0 + s * 16 + g * 8 + j;
        S[(size_t)row * SEQ + n0 + t * 16 + nl] = acc[s][t][j] * scale;
      }
}

// ---- Kernel 3: row softmax; write bf16 P in place over the f32 row ----------
__global__ void __launch_bounds__(256)
softmax_kernel(float* __restrict__ S) {
  __shared__ float ex[SEQ];
  __shared__ float red[256];
  const int tid = threadIdx.x;
  float* row = S + (size_t)blockIdx.x * SEQ;

  float m = -3.4e38f;
  for (int j = tid; j < SEQ; j += 256) {
    const float v = row[j];
    ex[j] = v;
    m = fmaxf(m, v);
  }
  red[tid] = m;
  __syncthreads();
#pragma unroll
  for (int s = 128; s > 0; s >>= 1) {
    if (tid < s) red[tid] = fmaxf(red[tid], red[tid + s]);
    __syncthreads();
  }
  m = red[0];
  __syncthreads();

  float sum = 0.f;
  for (int j = tid; j < SEQ; j += 256) {
    const float e = __expf(ex[j] - m);
    ex[j] = e;
    sum += e;
  }
  red[tid] = sum;
  __syncthreads();
#pragma unroll
  for (int s = 128; s > 0; s >>= 1) {
    if (tid < s) red[tid] += red[tid + s];
    __syncthreads();
  }
  const float inv = 1.f / red[0];

  bf16_t* P = (bf16_t*)row;                    // consumer pitch = 2*SEQ bf16
  for (int j = tid; j < SEQ; j += 256) P[j] = (bf16_t)(ex[j] * inv);
}

// ---- Kernel 4: out = P*V for one batch (bf16 x bf16 -> f32) -----------------
// Block tile 128x128, async LDS double-buffering like kernel 1 (bf16 tiles).
__global__ void __launch_bounds__(256)
out_gemm_kernel(const bf16_t* __restrict__ P, const bf16_t* __restrict__ qkv_b,
                float* __restrict__ O) {
  __shared__ bf16_t As[2][128 * 32];  // [q][k], 8KB each
  __shared__ bf16_t Bs[2][32 * 128];  // [k][d], 8KB each

  const int tid   = threadIdx.x;
  const int lane  = tid & 31;
  const int wave  = tid >> 5;
  const int waveM = wave >> 1;
  const int waveN = wave & 1;
  const int g     = lane >> 4;
  const int nl    = lane & 15;

  const int nBlkN = HID / 128;        // 8
  const int q0 = (blockIdx.x / nBlkN) * 128;
  const int d0 = (blockIdx.x % nBlkN) * 128;

  const bf16_t* V = qkv_b + 2 * HID;  // cols [2048,3072), pitch 3072

  auto stage = [&](int buf, int k0) {
#pragma unroll
    for (int r = 0; r < 2; ++r) {     // A: 512 16B chunks (2/thread)
      const int c = tid + r * 256;
      const int row = c >> 2, koff = (c & 3) * 8;
      async_copy16(P + (size_t)(q0 + row) * (2 * SEQ) + k0 + koff,
                   &As[buf][row * 32 + koff]);
    }
#pragma unroll
    for (int r = 0; r < 2; ++r) {     // B: 512 16B chunks (2/thread)
      const int c = tid + r * 256;
      const int kr = c >> 4, noff = (c & 15) * 8;
      async_copy16(V + (size_t)(k0 + kr) * NQKV + d0 + noff,
                   &Bs[buf][kr * 128 + noff]);
    }
  };

  v8f acc[2][4];
#pragma unroll
  for (int s = 0; s < 2; ++s)
#pragma unroll
    for (int t = 0; t < 4; ++t) acc[s][t] = v8f{};

  stage(0, 0);
  int cur = 0;
  for (int kt = 0; kt < SEQ / 32; ++kt) {
    wait_async0();
    __syncthreads();
    if (kt + 1 < SEQ / 32) stage(cur ^ 1, (kt + 1) * 32);

    const bf16_t* Ab = &As[cur][0];
    const bf16_t* Bb = &Bs[cur][0];
    v16bf A0 = lds_a_bf16(Ab, waveM * 32 + nl, g);
    v16bf A1 = lds_a_bf16(Ab, waveM * 32 + 16 + nl, g);
    v16bf Bt[4];
#pragma unroll
    for (int t = 0; t < 4; ++t) Bt[t] = lds_b_bf16(Bb, waveN * 64 + t * 16 + nl, g);
#pragma unroll
    for (int t = 0; t < 4; ++t) {
      acc[0][t] = wmma_bf16(A0, Bt[t], acc[0][t]);
      acc[1][t] = wmma_bf16(A1, Bt[t], acc[1][t]);
    }
    cur ^= 1;
  }

  const int qw = q0 + waveM * 32;
  const int dw = d0 + waveN * 64;
#pragma unroll
  for (int s = 0; s < 2; ++s)
#pragma unroll
    for (int t = 0; t < 4; ++t)
#pragma unroll
      for (int j = 0; j < 8; ++j) {
        const int row = qw + s * 16 + g * 8 + j;
        O[(size_t)row * HID + dw + t * 16 + nl] = acc[s][t][j];
      }
}

extern "C" void kernel_launch(void* const* d_in, const int* in_sizes, int n_in,
                              void* d_out, int out_size, void* d_ws, size_t ws_size,
                              hipStream_t stream) {
  (void)in_sizes; (void)n_in; (void)out_size; (void)ws_size;
  const float* X    = (const float*)d_in[0];   // [4,2048,1024]
  const float* W    = (const float*)d_in[1];   // [1024,3072]
  const float* bias = (const float*)d_in[2];   // [3072]
  float* out        = (float*)d_out;           // [4,2048,1024]

  bf16_t* qkv = (bf16_t*)d_ws;                                   // 48 MB
  float*  S   = (float*)((char*)d_ws + (size_t)8192 * NQKV * 2); // 16 MB

  // (1) QKV projection: 64x24 blocks of 128x128.
  qkv_gemm_kernel<<<1536, 256, 0, stream>>>(X, W, bias, qkv);

  // (2)-(4) per batch, reusing the single score buffer (stream-ordered).
  for (int b = 0; b < 4; ++b) {
    const bf16_t* qkv_b = qkv + (size_t)b * SEQ * NQKV;
    scores_kernel<<<256, 256, 0, stream>>>(qkv_b, S);
    softmax_kernel<<<SEQ, 256, 0, stream>>>(S);
    out_gemm_kernel<<<128, 256, 0, stream>>>((const bf16_t*)S, qkv_b,
                                             out + (size_t)b * SEQ * HID);
  }
}
